// SlidingWindowAttention_17927193493835
// MI455X (gfx1250) — matrix operand order repaired and checked by reading, not compiled, not run on hardware
//
#include <hip/hip_runtime.h>
#include <hip/hip_bf16.h>

// ---------------- problem constants ----------------
#define SEQ    4096
#define HID    2048
#define NH     16
#define HD     128
#define WINH   256          // WINDOW_SIZE/2
#define NGLOB  16
#define SCALE  0.08838834764831843f   // 128^-0.5

typedef __bf16 bf16;
typedef __attribute__((ext_vector_type(16))) __bf16 bf16x16;
typedef __attribute__((ext_vector_type(8)))  float  f32x8;

#define WMMA_BF16(A, B, C) \
    __builtin_amdgcn_wmma_f32_16x16x32_bf16(false, (A), false, (B), (short)0, (C), false, false)

union AFrag { unsigned u[8]; bf16x16 v; };
union BFrag { uint4   q[2]; bf16x16 v; };

__device__ __forceinline__ void store_out(float* p, float v) { *p = v; }
__device__ __forceinline__ void store_out(bf16*  p, float v) { *p = (bf16)v; }

// CDNA5 async global->LDS copy, 16B per lane, tracked by ASYNCcnt.
__device__ __forceinline__ void async_copy_b128(const void* lds_dst, const void* gsrc) {
    unsigned ldsa = (unsigned)(size_t)lds_dst;                  // low 32 bits = LDS offset
    unsigned long long ga = (unsigned long long)(size_t)gsrc;   // global VA
    asm volatile("global_load_async_to_lds_b128 %0, %1, off"
                 :: "v"(ldsa), "v"(ga) : "memory");
}
__device__ __forceinline__ void wait_async0() {
    asm volatile("s_wait_asynccnt 0x0" ::: "memory");
}

// ---------------- fp32 -> bf16 convert ----------------
__global__ void swa_cvt_bf16(const float* __restrict__ in, bf16* __restrict__ out, int n) {
    int i = blockIdx.x * blockDim.x + threadIdx.x;
    int stride = gridDim.x * blockDim.x;
    for (; i < n; i += stride) out[i] = (bf16)in[i];
}

// ---------------- bf16 GEMM: C[M,N] = A[M,K] * B[K,N] ----------------
// 256 threads (8 waves), tile 128x128, K-step 32, double-buffered LDS.
// A tile staged with async global->LDS; B tile transposed via VGPR path.
template <typename OutT>
__global__ __launch_bounds__(256)
void swa_gemm_bf16(const bf16* __restrict__ A, const bf16* __restrict__ B,
                   OutT* __restrict__ C, int M, int N, int K) {
    constexpr int TM = 128, TN = 128, TK = 32, LSTR = TK + 8; // pad 8 elems
    __shared__ __align__(16) bf16 Alds[2][TM * LSTR];
    __shared__ __align__(16) bf16 Blds[2][TN * LSTR];         // transposed: [n][k]

    const int tid  = threadIdx.x;
    const int lane = tid & 31, wave = tid >> 5;
    const int wm = wave >> 1, wn = wave & 1;                  // 4 x 2 wave grid
    const int l16 = lane & 15, lh = lane >> 4;
    const int m0 = blockIdx.y * TM, n0 = blockIdx.x * TN;

    const int am = tid >> 1, ah = tid & 1;                    // A staging slot
    const int bkk = tid >> 3, bnb = (tid & 7) * 16;           // B staging slot

    f32x8 acc[2][4];
#pragma unroll
    for (int i = 0; i < 2; ++i)
#pragma unroll
        for (int j = 0; j < 4; ++j)
#pragma unroll
            for (int e = 0; e < 8; ++e) acc[i][j][e] = 0.0f;

    uint4 bx0, bx1;
    auto stageA_async = [&](int buf, int kb) {
        const bf16* src = A + (size_t)(m0 + am) * K + kb + ah * 16;
        const bf16* dst = &Alds[buf][am * LSTR + ah * 16];
        async_copy_b128(dst, src);
        async_copy_b128((const char*)dst + 16, (const char*)src + 16);
    };
    auto loadB = [&](int kb) {
        const bf16* src = B + (size_t)(kb + bkk) * N + n0 + bnb;
        bx0 = ((const uint4*)src)[0];
        bx1 = ((const uint4*)src)[1];
    };
    auto storeB = [&](int buf) {
        const bf16* e0 = (const bf16*)&bx0;
        const bf16* e1 = (const bf16*)&bx1;
#pragma unroll
        for (int j = 0; j < 8; ++j) Blds[buf][(bnb + j) * LSTR + bkk]     = e0[j];
#pragma unroll
        for (int j = 0; j < 8; ++j) Blds[buf][(bnb + 8 + j) * LSTR + bkk] = e1[j];
    };

    // prologue: stage tile 0 into buffer 0
    stageA_async(0, 0);
    loadB(0);
    storeB(0);
    wait_async0();
    __syncthreads();

    const int nsteps = K / TK;
    for (int s = 0; s < nsteps; ++s) {
        const int cur = s & 1, nxt = cur ^ 1;
        const bool has_next = (s + 1) < nsteps;
        if (has_next) {
            stageA_async(nxt, (s + 1) * TK);     // overlap with WMMAs below
            loadB((s + 1) * TK);
        }

        // ---- fragments from current buffers + WMMAs ----
        AFrag af[2];
#pragma unroll
        for (int mi = 0; mi < 2; ++mi) {
            int m = wm * 32 + mi * 16 + l16;
#pragma unroll
            for (int p = 0; p < 8; ++p) {
                int k = lh * 16 + p * 2;
                af[mi].u[p] = *(const unsigned*)(&Alds[cur][m * LSTR + k]);
            }
        }
        BFrag bfv[4];
#pragma unroll
        for (int ni = 0; ni < 4; ++ni) {
            int n = wn * 64 + ni * 16 + l16;
            const uint4* p = (const uint4*)(&Blds[cur][n * LSTR + lh * 16]);
            bfv[ni].q[0] = p[0]; bfv[ni].q[1] = p[1];
        }
#pragma unroll
        for (int mi = 0; mi < 2; ++mi)
#pragma unroll
            for (int ni = 0; ni < 4; ++ni)
                acc[mi][ni] = WMMA_BF16(af[mi].v, bfv[ni].v, acc[mi][ni]);

        if (has_next) {
            storeB(nxt);
            wait_async0();
            __syncthreads();
        }
    }

    // ---- epilogue: C layout -> global ----
#pragma unroll
    for (int mi = 0; mi < 2; ++mi)
#pragma unroll
        for (int ni = 0; ni < 4; ++ni) {
            int gn = n0 + wn * 64 + ni * 16 + l16;
#pragma unroll
            for (int r = 0; r < 8; ++r) {
                int gm = m0 + wm * 32 + mi * 16 + r + lh * 8;
                store_out(C + (size_t)gm * N + gn, acc[mi][ni][r]);
            }
        }
}

// ---------------- sliding-window flash attention ----------------
// grid = (NH, SEQ/128), 256 threads. Each wave owns 16 query rows.
__global__ __launch_bounds__(256)
void swa_attn(const bf16* __restrict__ Q, const bf16* __restrict__ Kmat,
              const bf16* __restrict__ V, bf16* __restrict__ O) {
    constexpr int KT   = 32;        // key tile
    constexpr int KSTR = HD + 8;    // Klds row stride (136)
    constexpr int VSTR = KT + 8;    // Vt / P row stride (40)
    __shared__ __align__(16) bf16 Klds[KT * KSTR];     // [key][dim]
    __shared__ __align__(16) bf16 Vt[HD * VSTR];       // [dim][key]
    __shared__ __align__(16) bf16 Pl[8 * 16 * VSTR];   // per-wave P patch

    const int head = blockIdx.x;
    const int q0   = blockIdx.y * 128;
    const int tid  = threadIdx.x, lane = tid & 31, wave = tid >> 5;
    const int l16 = lane & 15, lh = lane >> 4;
    const int qrow0 = q0 + wave * 16;
    const size_t hoff = (size_t)head * HD;

    // ---- load Q fragments (A-layout), 4 blocks over dim 128 ----
    AFrag qf[4];
    {
        const bf16* qrow = Q + (size_t)(qrow0 + l16) * HID + hoff;
#pragma unroll
        for (int t = 0; t < 4; ++t)
#pragma unroll
            for (int p = 0; p < 8; ++p) {
                int k = t * 32 + lh * 16 + p * 2;
                qf[t].u[p] = *(const unsigned*)(qrow + k);
            }
    }

    f32x8 o[8];
#pragma unroll
    for (int f = 0; f < 8; ++f)
#pragma unroll
        for (int e = 0; e < 8; ++e) o[f][e] = 0.0f;
    float mi[8], li[8];
#pragma unroll
    for (int r = 0; r < 8; ++r) { mi[r] = -__builtin_inff(); li[r] = 0.0f; }

    const int kk = tid >> 3, nb = (tid & 7) * 16;   // staging slot

    for (int kb = 0; kb < SEQ; kb += KT) {
        // block-uniform tile activity (sliding window + globals)
        bool active = (kb < NGLOB) || (q0 == 0) ||
                      ((kb + KT - 1 >= q0 - (WINH - 1)) && (kb <= q0 + 127 + (WINH - 1)));
        if (!active) continue;

        __syncthreads();
        // ---- K tile: async row copy to LDS; V tile: VGPR transpose ----
        {
            const bf16* ksrc = Kmat + (size_t)(kb + kk) * HID + hoff + nb;
            const bf16* kdst = &Klds[kk * KSTR + nb];
            async_copy_b128(kdst, ksrc);
            async_copy_b128((const char*)kdst + 16, (const char*)ksrc + 16);

            const bf16* vsrc = V + (size_t)(kb + kk) * HID + hoff + nb;
            uint4 v0 = ((const uint4*)vsrc)[0];
            uint4 v1 = ((const uint4*)vsrc)[1];
            const bf16* e0 = (const bf16*)&v0;
            const bf16* e1 = (const bf16*)&v1;
#pragma unroll
            for (int j = 0; j < 8; ++j) Vt[(nb + j) * VSTR + kk]     = e0[j];
#pragma unroll
            for (int j = 0; j < 8; ++j) Vt[(nb + 8 + j) * VSTR + kk] = e1[j];

            // hint the next tile's V rows toward the caches
            if (kb + KT < SEQ)
                __builtin_prefetch(V + (size_t)(kb + KT + kk) * HID + hoff + nb, 0, 1);
        }
        wait_async0();
        __syncthreads();

        // ---- S = Q * K^T : two 16x16 fragments over this 32-key tile ----
        f32x8 s[2];
#pragma unroll
        for (int h = 0; h < 2; ++h) {
#pragma unroll
            for (int e = 0; e < 8; ++e) s[h][e] = 0.0f;
            int key = h * 16 + l16;
#pragma unroll
            for (int t = 0; t < 4; ++t) {
                BFrag kf;
                const uint4* p = (const uint4*)(&Klds[key * KSTR + t * 32 + lh * 16]);
                kf.q[0] = p[0]; kf.q[1] = p[1];
                s[h] = WMMA_BF16(qf[t].v, kf.v, s[h]);
            }
        }

        // ---- mask + scale ----
#pragma unroll
        for (int h = 0; h < 2; ++h) {
            int kpos = kb + h * 16 + l16;
#pragma unroll
            for (int r = 0; r < 8; ++r) {
                int qpos = qrow0 + r + lh * 8;
                int d = qpos - kpos;
                bool keep = (d < WINH && d > -WINH) || (qpos < NGLOB) || (kpos < NGLOB);
                s[h][r] = keep ? (s[h][r] * SCALE) : -__builtin_inff();
            }
        }

        // ---- online softmax: rows live per-VGPR, columns across 16 lanes ----
#pragma unroll
        for (int r = 0; r < 8; ++r) {
            float rowmax = fmaxf(s[0][r], s[1][r]);
#pragma unroll
            for (int off = 1; off < 16; off <<= 1)
                rowmax = fmaxf(rowmax, __shfl_xor(rowmax, off, 32));
            float mnew = fmaxf(mi[r], rowmax);
            bool dead = (mnew == -__builtin_inff());
            float alpha = dead ? 1.0f : __expf(mi[r] - mnew);
            float p0 = dead ? 0.0f : __expf(s[0][r] - mnew);
            float p1 = dead ? 0.0f : __expf(s[1][r] - mnew);
            s[0][r] = p0; s[1][r] = p1;
            float rs = p0 + p1;
#pragma unroll
            for (int off = 1; off < 16; off <<= 1)
                rs += __shfl_xor(rs, off, 32);
            li[r] = li[r] * alpha + rs;
            mi[r] = mnew;
#pragma unroll
            for (int f = 0; f < 8; ++f) o[f][r] *= alpha;
        }

        // ---- route P: C-layout -> A-layout via wave-private LDS patch ----
        bf16* pw = Pl + wave * 16 * VSTR;
#pragma unroll
        for (int h = 0; h < 2; ++h)
#pragma unroll
            for (int r = 0; r < 8; ++r) {
                int m = r + lh * 8, n = h * 16 + l16;
                pw[m * VSTR + n] = (bf16)s[h][r];
            }
        AFrag pf;
#pragma unroll
        for (int p = 0; p < 8; ++p) {
            int k = lh * 16 + p * 2;
            pf.u[p] = *(const unsigned*)(pw + l16 * VSTR + k);
        }

        // ---- O += P * V : 8 dim-blocks of 16 ----
#pragma unroll
        for (int f = 0; f < 8; ++f) {
            BFrag vf;
            int dim = f * 16 + l16;
            const uint4* p = (const uint4*)(&Vt[dim * VSTR + lh * 16]);
            vf.q[0] = p[0]; vf.q[1] = p[1];
            o[f] = WMMA_BF16(pf.v, vf.v, o[f]);
        }
    }

    // ---- normalize and store (bf16, [seq][head*HD+dim]) ----
#pragma unroll
    for (int r = 0; r < 8; ++r) {
        float inv = 1.0f / li[r];
#pragma unroll
        for (int f = 0; f < 8; ++f) o[f][r] *= inv;
    }
#pragma unroll
    for (int f = 0; f < 8; ++f) {
        int dim = f * 16 + l16;
#pragma unroll
        for (int r = 0; r < 8; ++r) {
            int m = qrow0 + r + lh * 8;
            O[(size_t)m * HID + hoff + dim] = (bf16)o[f][r];
        }
    }
}

// ---------------- host launch ----------------
extern "C" void kernel_launch(void* const* d_in, const int* in_sizes, int n_in,
                              void* d_out, int out_size, void* d_ws, size_t ws_size,
                              hipStream_t stream) {
    const float* X  = (const float*)d_in[0];
    const float* Wq = (const float*)d_in[1];
    const float* Wk = (const float*)d_in[2];
    const float* Wv = (const float*)d_in[3];
    const float* Wo = (const float*)d_in[4];
    float* out = (float*)d_out;

    const size_t NX = (size_t)SEQ * HID;   // 8388608
    const size_t NW = (size_t)HID * HID;   // 4194304

    bf16* Xbf  = (bf16*)d_ws;
    bf16* Wqb  = Xbf  + NX;
    bf16* Wkb  = Wqb  + NW;
    bf16* Wvb  = Wkb  + NW;
    bf16* Wob  = Wvb  + NW;
    bf16* Qbf  = Wob  + NW;
    bf16* Kbf  = Qbf  + NX;
    bf16* Vbf  = Kbf  + NX;
    bf16* Abf  = Vbf  + NX;   // attention output (bf16)

    swa_cvt_bf16<<<2048, 256, 0, stream>>>(X,  Xbf, (int)NX);
    swa_cvt_bf16<<<2048, 256, 0, stream>>>(Wq, Wqb, (int)NW);
    swa_cvt_bf16<<<2048, 256, 0, stream>>>(Wk, Wkb, (int)NW);
    swa_cvt_bf16<<<2048, 256, 0, stream>>>(Wv, Wvb, (int)NW);
    swa_cvt_bf16<<<2048, 256, 0, stream>>>(Wo, Wob, (int)NW);

    dim3 gg(HID / 128, SEQ / 128);   // (16, 32)
    swa_gemm_bf16<bf16><<<gg, 256, 0, stream>>>(Xbf, Wqb, Qbf, SEQ, HID, HID);
    swa_gemm_bf16<bf16><<<gg, 256, 0, stream>>>(Xbf, Wkb, Kbf, SEQ, HID, HID);
    swa_gemm_bf16<bf16><<<gg, 256, 0, stream>>>(Xbf, Wvb, Vbf, SEQ, HID, HID);

    dim3 ga(NH, SEQ / 128);          // (16, 32)
    swa_attn<<<ga, 256, 0, stream>>>(Qbf, Kbf, Vbf, Abf);

    swa_gemm_bf16<float><<<gg, 256, 0, stream>>>(Abf, Wob, out, SEQ, HID, HID);
}
